// BeplerContactPredictor_23227183137347
// MI455X (gfx1250) — matrix operand order repaired and checked by reading, not compiled, not run on hardware
//
#include <hip/hip_runtime.h>
#include <hip/hip_bf16.h>
#include <stdint.h>

typedef __bf16 bf16_t;
typedef __attribute__((ext_vector_type(16))) __bf16 v16bf;
typedef __attribute__((ext_vector_type(2)))  __bf16 v2bf;
typedef __attribute__((ext_vector_type(8)))  float  v8f;

static const int LQ = 256;   // sequence length
static const int DQ = 256;   // embedding dim
static const int CH = 32;    // hidden channels

union FragBF {
    bf16_t   h[16];
    unsigned w[8];
    uint4    q[2];
    v16bf    v;
};

union BPair {
    unsigned u;
    v2bf     v;
    bf16_t   h[2];
};

// bf16-pair dot product with f32 accumulate (v_dot2_f32_bf16 if available)
__device__ __forceinline__ float dot2bf(unsigned hu, unsigned wu, float acc) {
#if __has_builtin(__builtin_amdgcn_fdot2_f32_bf16)
    BPair a; a.u = hu;
    BPair b; b.u = wu;
    return __builtin_amdgcn_fdot2_f32_bf16(a.v, b.v, acc, false);
#else
    acc = fmaf(__uint_as_float(hu << 16),
               __uint_as_float(wu << 16), acc);
    acc = fmaf(__uint_as_float(hu & 0xFFFF0000u),
               __uint_as_float(wu & 0xFFFF0000u), acc);
    return acc;
#endif
}

// ---------------------------------------------------------------------------
// Kernel 1: pairwise features (mul | absdiff) -> GEMM(K=512,N=32) via WMMA
//           -> +b1 -> ReLU -> 2D mask -> h stored as bf16 [B,L,L,32]
// One workgroup = (b, i, jblk of 128 j's). 8 waves, each wave: 16 j x 32 ch.
// ---------------------------------------------------------------------------
__global__ __launch_bounds__(256) void k_pair_gemm(
    const float* __restrict__ x,      // [B,L,D]
    const int*   __restrict__ plen,   // [B]
    const float* __restrict__ W1,     // [2D, 32]
    const float* __restrict__ b1,     // [32]
    bf16_t* __restrict__ hws)         // [B,L,L,32] bf16
{
    // W1 staged in LDS pre-swizzled into B-fragment order:
    // [ctile(2)][kc(16)][lane(32)][t(16)] bf16  = 32 KB
    __shared__ __align__(16) bf16_t sW1f[2 * 16 * 32 * 16];
    __shared__ __align__(16) float  sXi[DQ];

    const int tid  = threadIdx.x;
    const int bid  = blockIdx.x;
    const int jblk = bid & 1;
    const int i    = (bid >> 1) & (LQ - 1);
    const int b    = bid >> 9;

    // --- stage W1 (bf16, fragment-major) -----------------------------------
    for (int f = tid; f < 2 * 16 * 32 * 16; f += 256) {
        const int ctile = f >> 13;
        const int r     = f & 8191;
        const int kc    = r >> 9;
        const int r2    = r & 511;
        const int ln    = r2 >> 4;
        const int t     = r2 & 15;
        const int half  = ln >> 4;
        const int c     = (ln & 15) + ctile * 16;
        // chunks 0..7 -> mul region (rows 0..255), 8..15 -> absdiff (256..511)
        const int wbase = ((kc >= 8) ? DQ : 0) + (kc & 7) * 32;
        const int srow  = wbase + ((t < 8) ? (half * 8 + t) : (16 + half * 8 + (t - 8)));  // B: lanes 0-15 K 0..7,16..23, lanes 16-31 K 8..15,24..31
        sW1f[f] = (bf16_t)W1[srow * CH + c];
    }
    // --- stage x[b,i,:] -----------------------------------------------------
    sXi[tid] = x[((size_t)b * LQ + i) * DQ + tid];
    __syncthreads();

    const int wv   = tid >> 5;
    const int lane = tid & 31;
    const int half = lane >> 4;
    const int j0   = jblk * 128 + wv * 16;
    const int jr   = j0 + (lane & 15);
    const float* __restrict__ xj = x + ((size_t)b * LQ + jr) * DQ;

    // hint: pull this wave's x row toward the WGP (global_prefetch_b8)
    __builtin_prefetch(xj, 0, 3);

    v8f c0 = {};
    v8f c1 = {};

    // ---- region 1: elementwise product (K chunks 0..7) --------------------
#pragma unroll 2
    for (int kc = 0; kc < 8; ++kc) {
        const int dbase = kc * 32;
        const int o1 = dbase + half * 8;
        const int o2 = dbase + 16 + half * 8;

        float fj[16], fi[16];
        *(float4*)&fj[0]  = *(const float4*)(xj + o1);
        *(float4*)&fj[4]  = *(const float4*)(xj + o1 + 4);
        *(float4*)&fj[8]  = *(const float4*)(xj + o2);
        *(float4*)&fj[12] = *(const float4*)(xj + o2 + 4);
        *(float4*)&fi[0]  = *(const float4*)(sXi + o1);
        *(float4*)&fi[4]  = *(const float4*)(sXi + o1 + 4);
        *(float4*)&fi[8]  = *(const float4*)(sXi + o2);
        *(float4*)&fi[12] = *(const float4*)(sXi + o2 + 4);

        FragBF a;
#pragma unroll
        for (int t = 0; t < 16; ++t)
            a.h[t] = (bf16_t)(fi[t] * fj[t]);      // v_cvt_pk_bf16_f32

        FragBF bA, bB;
        const int bofs = (kc * 32 + lane) * 16;
        bA.q[0] = *(const uint4*)&sW1f[bofs];
        bA.q[1] = *(const uint4*)&sW1f[bofs + 8];
        bB.q[0] = *(const uint4*)&sW1f[8192 + bofs];
        bB.q[1] = *(const uint4*)&sW1f[8192 + bofs + 8];

        c0 = __builtin_amdgcn_wmma_f32_16x16x32_bf16(false, a.v, false, bA.v,
                                                     (short)0, c0, false, false);
        c1 = __builtin_amdgcn_wmma_f32_16x16x32_bf16(false, a.v, false, bB.v,
                                                     (short)0, c1, false, false);
    }

    // ---- region 2: |difference| (K chunks 8..15) --------------------------
    // abs is applied in the packed bf16 domain (sign-bit mask) to avoid the
    // unselectable packed-f32-fabs DAG: |RNE(x)| == RNE(|x|).
#pragma unroll 2
    for (int kc = 8; kc < 16; ++kc) {
        const int dbase = (kc - 8) * 32;
        const int o1 = dbase + half * 8;
        const int o2 = dbase + 16 + half * 8;

        float fj[16], fi[16];
        *(float4*)&fj[0]  = *(const float4*)(xj + o1);
        *(float4*)&fj[4]  = *(const float4*)(xj + o1 + 4);
        *(float4*)&fj[8]  = *(const float4*)(xj + o2);
        *(float4*)&fj[12] = *(const float4*)(xj + o2 + 4);
        *(float4*)&fi[0]  = *(const float4*)(sXi + o1);
        *(float4*)&fi[4]  = *(const float4*)(sXi + o1 + 4);
        *(float4*)&fi[8]  = *(const float4*)(sXi + o2);
        *(float4*)&fi[12] = *(const float4*)(sXi + o2 + 4);

        FragBF a;
#pragma unroll
        for (int t = 0; t < 16; ++t)
            a.h[t] = (bf16_t)(fi[t] - fj[t]);      // signed diff, packed cvt
#pragma unroll
        for (int t = 0; t < 8; ++t)
            a.w[t] &= 0x7FFF7FFFu;                 // clear bf16 sign bits

        FragBF bA, bB;
        const int bofs = (kc * 32 + lane) * 16;
        bA.q[0] = *(const uint4*)&sW1f[bofs];
        bA.q[1] = *(const uint4*)&sW1f[bofs + 8];
        bB.q[0] = *(const uint4*)&sW1f[8192 + bofs];
        bB.q[1] = *(const uint4*)&sW1f[8192 + bofs + 8];

        c0 = __builtin_amdgcn_wmma_f32_16x16x32_bf16(false, a.v, false, bA.v,
                                                     (short)0, c0, false, false);
        c1 = __builtin_amdgcn_wmma_f32_16x16x32_bf16(false, a.v, false, bB.v,
                                                     (short)0, c1, false, false);
    }

    // --- epilogue: bias, relu, mask, store bf16 h --------------------------
    const int lenb = plen[b];
    const int nc   = lane & 15;          // C/D layout: N = lane (per 16)
    const float bias0 = b1[nc];
    const float bias1 = b1[nc + 16];
    const bool iok = (i < lenb);
    bf16_t* __restrict__ hp = hws + (((size_t)b * LQ + i) * LQ) * CH;

#pragma unroll
    for (int r = 0; r < 8; ++r) {
        const int  jrow = j0 + r + half * 8;   // C/D: lanes 0-15 M=r, 16-31 M=r+8
        const bool ok   = iok && (jrow < lenb);
        float v0 = c0[r] + bias0; v0 = v0 > 0.f ? v0 : 0.f; if (!ok) v0 = 0.f;
        float v1 = c1[r] + bias1; v1 = v1 > 0.f ? v1 : 0.f; if (!ok) v1 = 0.f;
        hp[(size_t)jrow * CH + nc]      = (bf16_t)v0;
        hp[(size_t)jrow * CH + nc + 16] = (bf16_t)v1;
    }
}

// ---------------------------------------------------------------------------
// Kernel 2: 7x7 SAME conv, 32->1 channels, + b2, final 2D mask.
// 16x16 output tile per workgroup; halo tile 22x22x32 bf16 in LDS (31 KB).
// bf16-pair dot products: one v_dot2_f32_bf16 per 2 channels.
// ---------------------------------------------------------------------------
__global__ __launch_bounds__(256) void k_conv7(
    const unsigned short* __restrict__ hws,  // [B,L,L,32] bf16 bits
    const int*   __restrict__ plen,
    const float* __restrict__ W2,            // [7,7,32,1] HWIO
    const float* __restrict__ b2,            // [1]
    float* __restrict__ out)                 // [B,L,L,1]
{
    __shared__ __align__(16) unsigned short sh[22 * 22 * 32];
    __shared__ __align__(16) unsigned sW2p[49 * 16];   // packed bf16 weight pairs

    const int tid = threadIdx.x;
    const int bid = blockIdx.x;
    const int tj  = bid & 15;
    const int ti  = (bid >> 4) & 15;
    const int b   = bid >> 8;
    const int i0  = ti * 16;
    const int j0  = tj * 16;

    const unsigned short* __restrict__ hb = hws + (size_t)b * LQ * LQ * CH;

    for (int p = tid; p < 22 * 22; p += 256) {
        const int pi = p / 22, pj = p % 22;
        const int gi = i0 - 3 + pi, gj = j0 - 3 + pj;
        uint4 v0 = {0, 0, 0, 0}, v1 = v0, v2 = v0, v3 = v0;
        if (gi >= 0 && gi < LQ && gj >= 0 && gj < LQ) {
            const uint4* src = (const uint4*)(hb + ((size_t)gi * LQ + gj) * CH);
            v0 = src[0]; v1 = src[1]; v2 = src[2]; v3 = src[3];
        }
        uint4* dst = (uint4*)(sh + p * 32);
        dst[0] = v0; dst[1] = v1; dst[2] = v2; dst[3] = v3;
    }
    for (int t = tid; t < 49 * 16; t += 256) {
        BPair p;
        p.h[0] = (bf16_t)W2[2 * t];
        p.h[1] = (bf16_t)W2[2 * t + 1];
        sW2p[t] = p.u;
    }
    __syncthreads();

    const int li = tid >> 4, lj = tid & 15;
    float acc = 0.f;
#pragma unroll
    for (int di = 0; di < 7; ++di) {
#pragma unroll
        for (int dj = 0; dj < 7; ++dj) {
            const uint4* hq = (const uint4*)(sh + (((li + di) * 22) + (lj + dj)) * 32);
            const uint4* wq = (const uint4*)(sW2p + (di * 7 + dj) * 16);
#pragma unroll
            for (int cp = 0; cp < 4; ++cp) {     // 8 channels per step
                const uint4 u = hq[cp];
                const uint4 w = wq[cp];
                acc = dot2bf(u.x, w.x, acc);
                acc = dot2bf(u.y, w.y, acc);
                acc = dot2bf(u.z, w.z, acc);
                acc = dot2bf(u.w, w.w, acc);
            }
        }
    }

    const int gi = i0 + li, gj = j0 + lj;
    const int lenb = plen[b];
    float v = acc + b2[0];
    if (!((gi < lenb) && (gj < lenb))) v = 0.f;
    out[((size_t)b * LQ + gi) * LQ + gj] = v;
}

// ---------------------------------------------------------------------------
extern "C" void kernel_launch(void* const* d_in, const int* in_sizes, int n_in,
                              void* d_out, int out_size, void* d_ws, size_t ws_size,
                              hipStream_t stream)
{
    const float* x    = (const float*)d_in[0];   // [8,256,256] f32
    const int*   plen = (const int*)  d_in[1];   // [8] i32
    const float* W1   = (const float*)d_in[2];   // [512,32] f32
    const float* b1   = (const float*)d_in[3];   // [32] f32
    const float* W2   = (const float*)d_in[4];   // [7,7,32,1] f32
    const float* b2   = (const float*)d_in[5];   // [1] f32

    bf16_t* hws = (bf16_t*)d_ws; // needs 8*256*256*32*2 = 32 MB

    k_pair_gemm<<<dim3(8 * LQ * 2), dim3(256), 0, stream>>>(x, plen, W1, b1, hws);
    k_conv7<<<dim3(8 * 16 * 16), dim3(256), 0, stream>>>((const unsigned short*)hws,
                                                         plen, W2, b2, (float*)d_out);
}
